// LinkGNN_48619029790901
// MI455X (gfx1250) — matrix operand order repaired and checked by reading, not compiled, not run on hardware
//
#include <hip/hip_runtime.h>
#include <hip/hip_bf16.h>

typedef __attribute__((ext_vector_type(16))) _Float16 v16h;
typedef __attribute__((ext_vector_type(8)))  float    v8f;
typedef __attribute__((ext_vector_type(4)))  float    v4f;
typedef __attribute__((ext_vector_type(4)))  unsigned int v4u;

// GCC-style vector matching the async-LDS builtin's parameter type
typedef int v4i __attribute__((vector_size(16)));
typedef __attribute__((address_space(1))) v4i* v4i_gptr;
typedef __attribute__((address_space(3))) v4i* v4i_lptr;

#define N_NODES   100000
#define IN_DIM    256
#define HID       256
#define OUT_DIM   128
#define E_EDGES   500000
#define NODE_TILES 6250     // N_NODES / 16
#define EDGE_TILES 31250    // E_EDGES / 16 (per edge set)
#define WS1_BYTES  (256 * 256 * 2)   // packed f16 Ws1 = 131072 B

#if __has_builtin(__builtin_amdgcn_global_load_async_to_lds_b128)
#define HAVE_ASYNC_LDS 1
#else
#define HAVE_ASYNC_LDS 0
#endif

__device__ __forceinline__ void wait_async_zero() {
#if __has_builtin(__builtin_amdgcn_s_wait_asynccnt)
  __builtin_amdgcn_s_wait_asynccnt(0);
#else
  asm volatile("s_wait_asynccnt 0" ::: "memory");
#endif
}

// ---------------------------------------------------------------------------
// WMMA helpers (CDNA5 v_wmma_f32_16x16x32_f16, wave32)
// ---------------------------------------------------------------------------
__device__ __forceinline__ v8f wmma_f16(v16h a, v16h b, v8f c) {
  // (neg_a, A, neg_b, B, c_mod, C, reuse_a, reuse_b)
  return __builtin_amdgcn_wmma_f32_16x16x32_f16(false, a, false, b, (short)0, c,
                                                false, false);
}

// B fragments pre-packed: each lane's 16 f16 contiguous (32 B).
__device__ __forceinline__ v16h load_bfrag(const _Float16* __restrict__ base,
                                           int blk, int lane) {
  return *(const v16h*)(base + (((blk << 5) + lane) << 4));
}

__device__ __forceinline__ void load8f(const float* __restrict__ p, float* d) {
  v4f a = *(const v4f*)p;
  v4f b = *(const v4f*)(p + 4);
  d[0] = a.x; d[1] = a.y; d[2] = a.z; d[3] = a.w;
  d[4] = b.x; d[5] = b.y; d[6] = b.z; d[7] = b.w;
}

// A-matrix 16x32 f16 layout: lane l holds row m=l%16; elems 0..7 = K k0+8h..,
// elems 8..15 = K k0+16+8h.. (h = l>>4).
__device__ __forceinline__ v16h make_afrag_f32(const float* __restrict__ row,
                                               int k0, int half) {
  float t0[8], t1[8];
  load8f(row + k0 + 8 * half, t0);
  load8f(row + k0 + 16 + 8 * half, t1);
  v16h a;
#pragma unroll
  for (int j = 0; j < 8; ++j) {
    a[j]     = (_Float16)t0[j];
    a[8 + j] = (_Float16)t1[j];
  }
  return a;
}

__device__ __forceinline__ v16h make_afrag_lds(const _Float16* __restrict__ row,
                                               int k0, int half) {
  v16h a;
#pragma unroll
  for (int j = 0; j < 8; ++j) {
    a[j]     = row[k0 + 8 * half + j];
    a[8 + j] = row[k0 + 16 + 8 * half + j];
  }
  return a;
}

// Build the 8 pair-feature A fragments for one edge (hu,hv in fp32 global H).
// feats K [0,128) = |hu-hv| -> frags 0..3 ; [128,256) = hu*hv -> frags 4..7.
__device__ __forceinline__ void build_pair_afrags(const float* __restrict__ Hu,
                                                  const float* __restrict__ Hv,
                                                  int half, v16h* afr) {
#pragma unroll
  for (int s = 0; s < 4; ++s) {
    float au[16], av[16];
    load8f(Hu + 32 * s + 8 * half,      au);
    load8f(Hu + 32 * s + 16 + 8 * half, au + 8);
    load8f(Hv + 32 * s + 8 * half,      av);
    load8f(Hv + 32 * s + 16 + 8 * half, av + 8);
    v16h fa, fm;
#pragma unroll
    for (int j = 0; j < 16; ++j) {
      float d = au[j] - av[j];
      fa[j] = (_Float16)fabsf(d);
      fm[j] = (_Float16)(au[j] * av[j]);
    }
    afr[s]     = fa;
    afr[s + 4] = fm;
  }
}

// ---------------------------------------------------------------------------
// Weight pack: fp32 row-major W[K][Ncols] -> f16 WMMA B-fragment layout.
// ---------------------------------------------------------------------------
__global__ void pack_b_kernel(const float* __restrict__ W,
                              _Float16* __restrict__ outp,
                              int Ncols, int total) {
  int t = blockIdx.x * blockDim.x + threadIdx.x;
  if (t >= total) return;
  const int j  = t & 15;
  const int l  = (t >> 4) & 31;
  const int bi = t >> 9;
  const int NT = Ncols >> 4;
  const int nt = bi % NT;
  const int kt = bi / NT;
  const int n  = nt * 16 + (l & 15);
  const int k  = kt * 32 + ((l >> 4) << 3) + (j & 7) + ((j >> 3) << 4);
  outp[t] = (_Float16)W[k * Ncols + n];
}

// ---------------------------------------------------------------------------
// Node encoder: H = relu(x@Wn1+bn1)@Wn2+bn2  (one wave per 16-row tile)
// ---------------------------------------------------------------------------
__global__ void __launch_bounds__(64)
node_encoder_kernel(const float* __restrict__ x,
                    const _Float16* __restrict__ Wn1p,
                    const float* __restrict__ bn1,
                    const _Float16* __restrict__ Wn2p,
                    const float* __restrict__ bn2,
                    float* __restrict__ Hout) {
  __shared__ _Float16 lds[2][16 * HID];  // per-wave 16x256 f16 tile (8 KB each)
  const int wave = threadIdx.x >> 5;
  const int lane = threadIdx.x & 31;
  const int col  = lane & 15;
  const int half = lane >> 4;
  const int tile = blockIdx.x * 2 + wave;   // grid sized exactly: no guard
  const int row0 = tile * 16;

  // A fragments from x (fp32 -> f16 on the fly)
  v16h ax[8];
  const float* xrow = x + (size_t)(row0 + col) * IN_DIM;
#pragma unroll
  for (int kt = 0; kt < 8; ++kt) ax[kt] = make_afrag_f32(xrow, kt * 32, half);

  // Stage 1: [16x256] = relu(x_tile @ Wn1 + bn1) -> LDS (f16, row-major)
  _Float16* hm = lds[wave];
  for (int nt = 0; nt < 16; ++nt) {
    v8f acc = {};
#pragma unroll
    for (int kt = 0; kt < 8; ++kt)
      acc = wmma_f16(ax[kt], load_bfrag(Wn1p, kt * 16 + nt, lane), acc);
    const float b = bn1[nt * 16 + col];
#pragma unroll
    for (int r = 0; r < 8; ++r) {
      float v = acc[r] + b;
      v = v > 0.0f ? v : 0.0f;
      hm[(r + 8 * half) * HID + nt * 16 + col] = (_Float16)v;
    }
  }
  __syncthreads();  // D-layout writes -> A-layout reads (cross-lane)

  // Stage 2: H_tile = hm @ Wn2 + bn2 -> fp32 global
  v16h ah[8];
  const _Float16* hrow = hm + col * HID;
#pragma unroll
  for (int kt = 0; kt < 8; ++kt) ah[kt] = make_afrag_lds(hrow, kt * 32, half);

  for (int nt = 0; nt < 8; ++nt) {
    v8f acc = {};
#pragma unroll
    for (int kt = 0; kt < 8; ++kt)
      acc = wmma_f16(ah[kt], load_bfrag(Wn2p, kt * 8 + nt, lane), acc);
    const float b = bn2[nt * 16 + col];
#pragma unroll
    for (int r = 0; r < 8; ++r)
      Hout[(size_t)(row0 + r + 8 * half) * OUT_DIM + nt * 16 + col] = acc[r] + b;
  }
}

// ---------------------------------------------------------------------------
// Pair scorer: gather H rows, feats=[|hu-hv|, hu*hv], relu(feats@Ws1+bs1)@Ws2
// Block = 64 threads (2 waves); each wave handles 2 edge tiles (32 edges) so
// every B fragment (read from LDS) feeds two WMMAs. Packed Ws1 (128 KB) is
// staged to dynamic LDS once per block via async global->LDS copy.
// ---------------------------------------------------------------------------
__global__ void __launch_bounds__(64)
pair_score_kernel(const float* __restrict__ H,
                  const int* __restrict__ pu, const int* __restrict__ pv,
                  const int* __restrict__ nu, const int* __restrict__ nv,
                  const _Float16* __restrict__ Ws1p,
                  const float* __restrict__ bs1,
                  const float* __restrict__ Ws2,
                  const float* __restrict__ bs2,
                  float* __restrict__ out) {
  extern __shared__ _Float16 smem[];   // 131072 B: packed Ws1 staged per block

  // ---- stage packed Ws1 into LDS (64 threads x 128 x 16 B, interleaved) ----
#if HAVE_ASYNC_LDS
  {
    const char* gc = (const char*)Ws1p;
    char*       lc = (char*)smem;
#pragma unroll 4
    for (int i = 0; i < 128; ++i) {
      const int off = (i * 64 + threadIdx.x) * 16;
      __builtin_amdgcn_global_load_async_to_lds_b128(
          (v4i_gptr)(gc + off), (v4i_lptr)(lc + off), 0, 0);
    }
    wait_async_zero();
  }
#else
  {
    const v4u* g = (const v4u*)Ws1p;
    v4u* l = (v4u*)smem;
#pragma unroll 4
    for (int i = 0; i < 128; ++i) {
      const int off = i * 64 + threadIdx.x;
      l[off] = g[off];
    }
  }
#endif
  __syncthreads();

  const int wave = threadIdx.x >> 5;
  const int lane = threadIdx.x & 31;
  const int col  = lane & 15;
  const int half = lane >> 4;

  // pair index p covers global edge-tiles 2p, 2p+1 (set boundary 31250 is even)
  const int p = blockIdx.x * 2 + wave;        // 0..31249, grid sized exactly
  int t0 = p * 2;
  const int* U; const int* V; float* O;
  if (t0 < EDGE_TILES) { U = pu; V = pv; O = out; }
  else                 { U = nu; V = nv; O = out + E_EDGES; t0 -= EDGE_TILES; }
  const int e0 = t0 * 16;                     // 32 consecutive edges

  // ---- gather + build A fragments for both tiles ----
  const int eA = e0 + col;
  const int eB = e0 + 16 + col;
  const int uA = U[eA], vA = V[eA];
  const int uB = U[eB], vB = V[eB];

  v16h af0[8], af1[8];
  build_pair_afrags(H + (size_t)uA * OUT_DIM, H + (size_t)vA * OUT_DIM, half, af0);
  build_pair_afrags(H + (size_t)uB * OUT_DIM, H + (size_t)vB * OUT_DIM, half, af1);

  float lp0[8], lp1[8];
#pragma unroll
  for (int r = 0; r < 8; ++r) { lp0[r] = 0.0f; lp1[r] = 0.0f; }

  for (int nt = 0; nt < 16; ++nt) {
    v8f a0 = {}, a1 = {};
#pragma unroll
    for (int kt = 0; kt < 8; ++kt) {
      const v16h b = load_bfrag(smem, kt * 16 + nt, lane);  // ds_load_b128 x2
      a0 = wmma_f16(af0[kt], b, a0);
      a1 = wmma_f16(af1[kt], b, a1);
    }
    const float bb = bs1[nt * 16 + col];
    const float w2 = Ws2[nt * 16 + col];  // Ws2 is [256,1]
#pragma unroll
    for (int r = 0; r < 8; ++r) {
      float h0 = a0[r] + bb; h0 = h0 > 0.0f ? h0 : 0.0f;
      float h1 = a1[r] + bb; h1 = h1 > 0.0f ? h1 : 0.0f;
      lp0[r] += h0 * w2;
      lp1[r] += h1 * w2;
    }
  }

  // Reduce within each 16-lane half (D-layout rows split across halves).
#pragma unroll
  for (int mask = 1; mask <= 8; mask <<= 1) {
#pragma unroll
    for (int r = 0; r < 8; ++r) {
      lp0[r] += __shfl_xor(lp0[r], mask, 32);
      lp1[r] += __shfl_xor(lp1[r], mask, 32);
    }
  }

  const float b2 = bs2[0];
  float my0 = 0.0f, my1 = 0.0f;
#pragma unroll
  for (int r = 0; r < 8; ++r) {
    if ((lane & 7) == r) { my0 = lp0[r]; my1 = lp1[r]; }
  }
  if ((lane & 8) == 0) {
    O[e0 +      8 * half + (lane & 7)] = my0 + b2;
    O[e0 + 16 + 8 * half + (lane & 7)] = my1 + b2;
  }
}

// ---------------------------------------------------------------------------
// Launch
// ---------------------------------------------------------------------------
extern "C" void kernel_launch(void* const* d_in, const int* in_sizes, int n_in,
                              void* d_out, int out_size, void* d_ws, size_t ws_size,
                              hipStream_t stream) {
  const float* x     = (const float*)d_in[0];
  const int*   pos_u = (const int*)d_in[1];
  const int*   pos_v = (const int*)d_in[2];
  const int*   neg_u = (const int*)d_in[3];
  const int*   neg_v = (const int*)d_in[4];
  const float* Wn1   = (const float*)d_in[5];
  const float* bn1   = (const float*)d_in[6];
  const float* Wn2   = (const float*)d_in[7];
  const float* bn2   = (const float*)d_in[8];
  const float* Ws1   = (const float*)d_in[9];
  const float* bs1   = (const float*)d_in[10];
  const float* Ws2   = (const float*)d_in[11];
  const float* bs2   = (const float*)d_in[12];

  // Workspace: packed f16 weights (320 KB total)
  _Float16* Wn1p = (_Float16*)d_ws;          // 256*256
  _Float16* Wn2p = Wn1p + 256 * 256;         // 256*128
  _Float16* Ws1p = Wn2p + 256 * 128;         // 256*256

  float* out  = (float*)d_out;               // [pos E | neg E | H N*128]
  float* Hout = out + 2 * E_EDGES;

  pack_b_kernel<<<(256 * 256) / 256, 256, 0, stream>>>(Wn1, Wn1p, 256, 256 * 256);
  pack_b_kernel<<<(256 * 128) / 256, 256, 0, stream>>>(Wn2, Wn2p, 128, 256 * 128);
  pack_b_kernel<<<(256 * 256) / 256, 256, 0, stream>>>(Ws1, Ws1p, 256, 256 * 256);

  node_encoder_kernel<<<NODE_TILES / 2, 64, 0, stream>>>(x, Wn1p, bn1, Wn2p, bn2, Hout);

  // 62500 edge tiles = 15625 blocks x 2 waves x 2 tiles
  pair_score_kernel<<<EDGE_TILES * 2 / 4, 64, WS1_BYTES, stream>>>(
      Hout, pos_u, pos_v, neg_u, neg_v, Ws1p, bs1, Ws2, bs2, out);
}
